// HetNetGNN_50044958933533
// MI455X (gfx1250) — compile-verified, matrix operand order checked by reading, not run on hardware
//
#include <hip/hip_runtime.h>
#include <hip/hip_bf16.h>

typedef __attribute__((ext_vector_type(16))) _Float16 v16h;
typedef __attribute__((ext_vector_type(8)))  float    v8f;

#define WPB 8   // waves per block (wave32); one wave == one sample

struct GnnParams {
  const float* x_ue; const float* x_ap; const float* ea_up; const float* ea_dn;
  // per power layer l: edge MLP, node-msg MLP, node-upd MLP, power head
  const float* eW1[2]; const float* eB1[2]; const float* eW2[2]; const float* eB2[2];
  const float* mW1[2]; const float* mB1[2]; const float* mW2[2]; const float* mB2[2];
  const float* uW1[2]; const float* uB1[2]; const float* uW2[2]; const float* uB2[2];
  const float* pW1[2]; const float* pB1[2]; const float* pW2[2]; const float* pB2[2];
  const float* aW1; const float* aB1; const float* aW2; const float* aB2;
  float* o_xue; float* o_eup;
  int S;
};

__global__ __launch_bounds__(WPB * 32)
void hetnet_fused(GnnParams P) {
  const int t = threadIdx.x & 31;          // lane
  const int w = threadIdx.x >> 5;          // wave in block
  int s = blockIdx.x * WPB + w;            // sample
  if (s >= P.S) s = P.S - 1;               // duplicate (identical) work; grid divides exactly anyway

  __shared__ float s_buf[WPB][32][33];     // AP-msg staging / per-UE channel sums
  __shared__ float s_apsum[WPB][2][32];    // per-sample AP message sums, per layer
  __shared__ float s_pw[WPB][16];          // final powers per UE

  // ---- Stage 0: per-AP node-message MLP (lane == AP), reduce over sample ----
  const float xap = P.x_ap[s * 32 + t];
  #pragma unroll
  for (int l = 0; l < 2; ++l) {
    float hh[16];
    #pragma unroll
    for (int j = 0; j < 16; ++j) hh[j] = fmaxf(fmaf(xap, P.mW1[l][j], P.mB1[l][j]), 0.f);
    #pragma unroll 4
    for (int ch = 0; ch < 32; ++ch) {
      float acc = P.mB2[l][ch];
      #pragma unroll
      for (int j = 0; j < 16; ++j) acc = fmaf(hh[j], P.mW2[l][j * 32 + ch], acc);
      s_buf[w][t][ch] = fmaxf(acc, 0.f);
    }
    __syncthreads();
    float ssum = 0.f;
    #pragma unroll 8
    for (int a = 0; a < 32; ++a) ssum += s_buf[w][a][t];
    s_apsum[w][l][t] = ssum;
    __syncthreads();
  }

  // ---- Stages 1+2: two PowerConv layers. Edge MLP second layer on WMMA. ----
  // A frag: M=16 edges (a tile), K=16 hidden zero-padded to 32. Lane<16 -> j 0..7, lane>=16 -> j 8..15.
  const int cn = t & 15;
  const int jb = (t < 16) ? 0 : 8;
  float pwprev = 0.f;

  #pragma unroll
  for (int l = 0; l < 2; ++l) {
    v16h Blo, Bhi;                          // B: K(32,zero-padded)xN(16), two N-halves
    float w1r[8], b1r[8];
    #pragma unroll
    for (int i = 0; i < 8; ++i) {
      Blo[i]     = (_Float16)P.eW2[l][(jb + i) * 32 + cn];
      Bhi[i]     = (_Float16)P.eW2[l][(jb + i) * 32 + 16 + cn];
      Blo[i + 8] = (_Float16)0.f;
      Bhi[i + 8] = (_Float16)0.f;
      w1r[i] = P.eW1[l][jb + i];
      b1r[i] = P.eB1[l][jb + i];
    }
    v8f cbLo, cbHi;                         // preload output bias into C operand
    const float bLo = P.eB2[l][cn], bHi = P.eB2[l][16 + cn];
    #pragma unroll
    for (int r = 0; r < 8; ++r) { cbLo[r] = bLo; cbHi[r] = bHi; }

    const float* edn = P.ea_dn + (size_t)s * 16 * 32 * 2;
    #pragma unroll 1
    for (int u = 0; u < 16; ++u) {
      if (u + 1 < 16) __builtin_prefetch(edn + ((u + 1) * 32 + (t & 15)) * 2, 0, 1);
      float zlo = 0.f, zhi = 0.f;
      #pragma unroll
      for (int half = 0; half < 2; ++half) { // a = 0..15 then 16..31
        const int a = (t & 15) + half * 16;
        const float x = edn[(u * 32 + a) * 2];   // ea_dn[:, 0]
        v16h A;
        #pragma unroll
        for (int i = 0; i < 8; ++i) {
          A[i]     = (_Float16)fmaxf(fmaf(x, w1r[i], b1r[i]), 0.f);
          A[i + 8] = (_Float16)0.f;
        }
        v8f dlo = __builtin_amdgcn_wmma_f32_16x16x32_f16(false, A, false, Blo, (short)0, cbLo, false, false);
        v8f dhi = __builtin_amdgcn_wmma_f32_16x16x32_f16(false, A, false, Bhi, (short)0, cbHi, false, false);
        #pragma unroll
        for (int r = 0; r < 8; ++r) {       // per-edge ReLU, then sum rows (edges)
          zlo += fmaxf(dlo[r], 0.f);
          zhi += fmaxf(dhi[r], 0.f);
        }
      }
      zlo += __shfl_xor(zlo, 16, 32);       // rows 0-7 + rows 8-15 of column cn
      zhi += __shfl_xor(zhi, 16, 32);
      if (t < 16) { s_buf[w][u][cn] = zlo; s_buf[w][u][16 + cn] = zhi; }
    }
    __syncthreads();

    // finalize: lane u < 16 -> mean, residual MLP, head, sigmoid
    if (t < 16) {
      const int u = t;
      float val[32];
      #pragma unroll
      for (int ch = 0; ch < 32; ++ch)
        val[ch] = (s_buf[w][u][ch] + s_apsum[w][l][ch]) * (1.f / 32.f);
      const float xp = (l == 0) ? P.x_ue[(s * 16 + u) * 2 + 1] : pwprev;
      float rh[16];
      #pragma unroll
      for (int j = 0; j < 16; ++j) rh[j] = fmaxf(fmaf(xp, P.uW1[l][j], P.uB1[l][j]), 0.f);
      #pragma unroll 4
      for (int ch = 0; ch < 32; ++ch) {
        float r = P.uB2[l][ch];
        #pragma unroll
        for (int j = 0; j < 16; ++j) r = fmaf(rh[j], P.uW2[l][j * 32 + ch], r);
        val[ch] += fmaxf(r, 0.f);
      }
      float logit = P.pB2[l][0];
      #pragma unroll 2
      for (int k = 0; k < 16; ++k) {
        float h = P.pB1[l][k];
        #pragma unroll
        for (int ch = 0; ch < 32; ++ch) h = fmaf(val[ch], P.pW1[l][ch * 16 + k], h);
        logit = fmaf(fmaxf(h, 0.f), P.pW2[l][k], logit);
      }
      pwprev = 1.f / (1.f + __expf(-logit));
    }
    __syncthreads();
  }

  if (t < 16) {
    const int g = s * 16 + t;
    P.o_xue[2 * g]     = P.x_ue[2 * g];     // p_max passes through
    P.o_xue[2 * g + 1] = pwprev;            // final power
    s_pw[w][t] = pwprev;
  }
  __syncthreads();

  // ---- Stage 3: EdgeConv ap-selection. sigmoid/softmax are monotone -> argmax of logit. ----
  for (int u = 0; u < 16; ++u) {
    const float pmax = P.x_ue[2 * (s * 16 + u)];
    const float pw   = s_pw[w][u];
    const int   e    = (s * 16 + u) * 32 + t;       // lane == AP within group
    const float2 ea2 = reinterpret_cast<const float2*>(P.ea_up)[e];
    float logit = P.aB2[0];
    #pragma unroll 4
    for (int k = 0; k < 16; ++k) {
      float h = P.aB1[k];
      h = fmaf(pmax,  P.aW1[0 * 16 + k], h);
      h = fmaf(pw,    P.aW1[1 * 16 + k], h);
      h = fmaf(xap,   P.aW1[2 * 16 + k], h);
      h = fmaf(ea2.x, P.aW1[3 * 16 + k], h);
      h = fmaf(ea2.y, P.aW1[4 * 16 + k], h);
      logit = fmaf(fmaxf(h, 0.f), P.aW2[k], logit);
    }
    float v = logit; int idx = t;                   // wave-wide argmax, first-max tie-break
    #pragma unroll
    for (int off = 16; off > 0; off >>= 1) {
      const float ov = __shfl_xor(v, off, 32);
      const int   oi = __shfl_xor(idx, off, 32);
      if (ov > v || (ov == v && oi < idx)) { v = ov; idx = oi; }
    }
    float2 o; o.x = ea2.x; o.y = (t == idx) ? 1.f : 0.f;
    reinterpret_cast<float2*>(P.o_eup)[e] = o;
  }
}

extern "C" void kernel_launch(void* const* d_in, const int* in_sizes, int n_in,
                              void* d_out, int out_size, void* d_ws, size_t ws_size,
                              hipStream_t stream) {
  (void)d_ws; (void)ws_size; (void)out_size;
  const float* x_ue  = (const float*)d_in[0];
  const float* x_ap  = (const float*)d_in[1];
  const float* ea_up = (const float*)d_in[2];
  const float* ea_dn = (const float*)d_in[3];
  const int N_UE2 = in_sizes[0];       // 2*N_UE
  const int N_AP  = in_sizes[1];
  const int E2    = in_sizes[2];       // 2*E
  const int S     = N_AP / 32;

  GnnParams P{};
  P.x_ue = x_ue; P.x_ap = x_ap; P.ea_up = ea_up; P.ea_dn = ea_dn; P.S = S;

  // Param leaves start at index 6. JAX pytree order sorts dict keys:
  // top: ap_sel < power; within a PowerConv: edge_dn, node_msg_ap, node_upd_ue, power_ue.
  // Fall back to insertion order if leaf sizes disagree (ap_sel W1 is [5,16] = 80).
  const bool sortedOrder = (n_in > 6 && in_sizes[6] == 80);
  int ap_base, pw_base[2], ge, gm, gu, gp;
  if (sortedOrder) { ap_base = 6;  pw_base[0] = 10; pw_base[1] = 26; ge = 0; gm = 1; gu = 2; gp = 3; }
  else             { pw_base[0] = 6; pw_base[1] = 22; ap_base = 38; gm = 0; ge = 1; gu = 2; gp = 3; }

  P.aW1 = (const float*)d_in[ap_base + 0]; P.aB1 = (const float*)d_in[ap_base + 1];
  P.aW2 = (const float*)d_in[ap_base + 2]; P.aB2 = (const float*)d_in[ap_base + 3];
  for (int l = 0; l < 2; ++l) {
    const int b = pw_base[l];
    P.eW1[l] = (const float*)d_in[b + 4 * ge + 0]; P.eB1[l] = (const float*)d_in[b + 4 * ge + 1];
    P.eW2[l] = (const float*)d_in[b + 4 * ge + 2]; P.eB2[l] = (const float*)d_in[b + 4 * ge + 3];
    P.mW1[l] = (const float*)d_in[b + 4 * gm + 0]; P.mB1[l] = (const float*)d_in[b + 4 * gm + 1];
    P.mW2[l] = (const float*)d_in[b + 4 * gm + 2]; P.mB2[l] = (const float*)d_in[b + 4 * gm + 3];
    P.uW1[l] = (const float*)d_in[b + 4 * gu + 0]; P.uB1[l] = (const float*)d_in[b + 4 * gu + 1];
    P.uW2[l] = (const float*)d_in[b + 4 * gu + 2]; P.uB2[l] = (const float*)d_in[b + 4 * gu + 3];
    P.pW1[l] = (const float*)d_in[b + 4 * gp + 0]; P.pB1[l] = (const float*)d_in[b + 4 * gp + 1];
    P.pW2[l] = (const float*)d_in[b + 4 * gp + 2]; P.pB2[l] = (const float*)d_in[b + 4 * gp + 3];
  }

  // Output layout: x_ue | x_ap | edge_attr_up | edge_attr_dn (flat, return order)
  float* out   = (float*)d_out;
  float* o_xue = out;
  float* o_xap = o_xue + N_UE2;
  float* o_eup = o_xap + N_AP;
  float* o_edn = o_eup + E2;
  P.o_xue = o_xue; P.o_eup = o_eup;

  // Pass-through outputs: stream-ordered D2D copies (graph-capture safe)
  hipMemcpyAsync(o_xap, (const void*)x_ap,  (size_t)N_AP * sizeof(float), hipMemcpyDeviceToDevice, stream);
  hipMemcpyAsync(o_edn, (const void*)ea_dn, (size_t)E2   * sizeof(float), hipMemcpyDeviceToDevice, stream);

  const int grid = (S + WPB - 1) / WPB;   // S=8192 -> 1024 blocks of 8 waves
  hetnet_fused<<<grid, WPB * 32, 0, stream>>>(P);
}
